// JointFluxSingleTransformerBlockControl_31155692765817
// MI455X (gfx1250) — matrix operand order repaired and checked by reading, not compiled, not run on hardware
//
#include <hip/hip_runtime.h>
#include <hip/hip_bf16.h>

// ---------------------------------------------------------------------------
// JointFluxSingleTransformerBlockControl for MI455X (gfx1250, wave32, WMMA)
// All heavy math via v_wmma_f32_16x16x32_bf16; fp32 weights converted to bf16
// in-register while staging to LDS (bandwidth-bound problem: ~800MB weights
// @23.3TB/s dominates; bf16 WMMA keeps compute below the memory floor).
// ---------------------------------------------------------------------------

typedef __attribute__((ext_vector_type(16))) __bf16 v16bf;
typedef __attribute__((ext_vector_type(8)))  __bf16 v8bf;
typedef __attribute__((ext_vector_type(8)))  float  v8f;

#define S_LEN 1024
#define DIMD  3072
#define MLPD  12288
#define CATD  15360
#define NHEAD 24
#define HDIM  128

__device__ __forceinline__ unsigned short f2bfu(float f) {
    unsigned u = __float_as_uint(f);
    return (unsigned short)((u + 0x7FFFu + ((u >> 16) & 1u)) >> 16);
}
__device__ __forceinline__ __bf16 f2bf(float f) {
    unsigned short s = f2bfu(f);
    return __builtin_bit_cast(__bf16, s);
}
__device__ __forceinline__ v16bf mkfrag(v8bf lo, v8bf hi) {
    v16bf r;
#pragma unroll
    for (int i = 0; i < 8; ++i) { r[i] = lo[i]; r[i + 8] = hi[i]; }
    return r;
}
__device__ __forceinline__ v8f wmma_bf16(v16bf a, v16bf b, v8f c) {
    return __builtin_amdgcn_wmma_f32_16x16x32_bf16(false, a, false, b,
                                                   (short)0, c, false, false);
}

// ---------------------------------------------------------------------------
// 1) silu(temb) for both streams
// ---------------------------------------------------------------------------
__global__ __launch_bounds__(256) void silu_kernel(const float* __restrict__ t0,
                                                   const float* __restrict__ t1,
                                                   float* __restrict__ out) {
    int i = blockIdx.x * 256 + threadIdx.x;
    if (i < DIMD)           { float x = t0[i];        out[i] = x / (1.f + __expf(-x)); }
    else if (i < 2 * DIMD)  { float x = t1[i - DIMD]; out[i] = x / (1.f + __expf(-x)); }
}

// ---------------------------------------------------------------------------
// 2) e = silu(temb) @ W^T + b   (GEMV, one wave per output row of W)
// ---------------------------------------------------------------------------
__global__ __launch_bounds__(256) void emb_gemv(const float* __restrict__ st,
                                                const float* __restrict__ W,
                                                const float* __restrict__ b,
                                                float* __restrict__ e) {
    int n = blockIdx.x * 8 + (threadIdx.x >> 5);
    int lane = threadIdx.x & 31;
    const float* w = W + (size_t)n * DIMD;
    float s = 0.f;
    for (int k = lane; k < DIMD; k += 32) s += st[k] * w[k];
#pragma unroll
    for (int m = 16; m > 0; m >>= 1) s += __shfl_xor(s, m, 32);
    if (lane == 0) e[n] = s + b[n];
}

// ---------------------------------------------------------------------------
// 3) adaLN-zero: layernorm(x)*(1+scale)+shift  -> bf16
// ---------------------------------------------------------------------------
__global__ __launch_bounds__(256) void adanorm_kernel(const float* __restrict__ X,
                                                      const float* __restrict__ e,
                                                      unsigned short* __restrict__ Y) {
    int row = blockIdx.x;
    const float* x = X + (size_t)row * DIMD;
    float vals[12];
    float s = 0.f, s2 = 0.f;
#pragma unroll
    for (int i = 0; i < 12; ++i) {
        float v = x[threadIdx.x + i * 256];
        vals[i] = v; s += v; s2 += v * v;
    }
#pragma unroll
    for (int m = 16; m > 0; m >>= 1) { s += __shfl_xor(s, m, 32); s2 += __shfl_xor(s2, m, 32); }
    __shared__ float sm[8], sv[8];
    int wid = threadIdx.x >> 5, lane = threadIdx.x & 31;
    if (lane == 0) { sm[wid] = s; sv[wid] = s2; }
    __syncthreads();
    float ts = 0.f, t2 = 0.f;
#pragma unroll
    for (int i = 0; i < 8; ++i) { ts += sm[i]; t2 += sv[i]; }
    float mu = ts / (float)DIMD;
    float var = t2 / (float)DIMD - mu * mu;
    float inv = rsqrtf(var + 1e-6f);
#pragma unroll
    for (int i = 0; i < 12; ++i) {
        int col = threadIdx.x + i * 256;
        float xh = (vals[i] - mu) * inv;
        float o = xh * (1.f + e[DIMD + col]) + e[col];
        Y[(size_t)row * DIMD + col] = f2bfu(o);
    }
}

// ---------------------------------------------------------------------------
// 4) Generic WMMA GEMM:  C[M,N] = A_bf16[M,K] * W_f32[N,K]^T (+bias, epilogue)
//    block = 256 thr (8 waves), tile 128x128, K-step 32.
//    mode 0: store f32  | 1: store bf16 | 2: gelu->bf16 | 3: resid+gate*(x)->f32
// ---------------------------------------------------------------------------
#define LDS_STRIDE 40   // 32 + 8 pad (keeps 16B alignment, avoids conflicts)

__global__ __launch_bounds__(256) void gemm_bf16(
    const unsigned short* __restrict__ A, int lda,
    const float* __restrict__ W, int ldw,
    const float* __restrict__ bias,
    float* __restrict__ Cf, unsigned short* __restrict__ Cb, int ldc,
    int K, int mode,
    const float* __restrict__ gate, const float* __restrict__ resid) {

    __shared__ __align__(16) unsigned short sA[128 * LDS_STRIDE];
    __shared__ __align__(16) unsigned short sW[128 * LDS_STRIDE];

    const int tid  = threadIdx.x;
    const int lane = tid & 31, wid = tid >> 5;
    const int half = lane >> 4, lr = lane & 15;
    const int m0 = blockIdx.y * 128, n0 = blockIdx.x * 128;
    const int wm = (wid & 1) * 64, wn = (wid >> 1) * 32;

    const int arow = tid >> 1;          // 0..127
    const int acol = (tid & 1) * 16;    // 0 or 16

    v8f acc[4][2];
#pragma unroll
    for (int mt = 0; mt < 4; ++mt)
#pragma unroll
        for (int nt = 0; nt < 2; ++nt) acc[mt][nt] = (v8f)0.f;

#pragma unroll 1
    for (int k0 = 0; k0 < K; k0 += 32) {
        // stage A (bf16 global -> LDS)
        const unsigned short* ga = A + (size_t)(m0 + arow) * lda + k0 + acol;
        v8bf a0 = *(const v8bf*)ga;
        v8bf a1 = *(const v8bf*)(ga + 8);
        *(v8bf*)&sA[arow * LDS_STRIDE + acol]     = a0;
        *(v8bf*)&sA[arow * LDS_STRIDE + acol + 8] = a1;
        // stage W (f32 global -> bf16 LDS)
        const float* gw = W + (size_t)(n0 + arow) * ldw + k0 + acol;
        float4 w0 = *(const float4*)(gw);
        float4 w1 = *(const float4*)(gw + 4);
        float4 w2 = *(const float4*)(gw + 8);
        float4 w3 = *(const float4*)(gw + 12);
        v8bf p0, p1;
        p0[0]=f2bf(w0.x); p0[1]=f2bf(w0.y); p0[2]=f2bf(w0.z); p0[3]=f2bf(w0.w);
        p0[4]=f2bf(w1.x); p0[5]=f2bf(w1.y); p0[6]=f2bf(w1.z); p0[7]=f2bf(w1.w);
        p1[0]=f2bf(w2.x); p1[1]=f2bf(w2.y); p1[2]=f2bf(w2.z); p1[3]=f2bf(w2.w);
        p1[4]=f2bf(w3.x); p1[5]=f2bf(w3.y); p1[6]=f2bf(w3.z); p1[7]=f2bf(w3.w);
        *(v8bf*)&sW[arow * LDS_STRIDE + acol]     = p0;
        *(v8bf*)&sW[arow * LDS_STRIDE + acol + 8] = p1;
        __syncthreads();

        v16bf af[4], bf_[2];
#pragma unroll
        for (int mt = 0; mt < 4; ++mt) {
            const unsigned short* pa = &sA[(wm + mt * 16 + lr) * LDS_STRIDE + half * 8];
            af[mt] = mkfrag(*(const v8bf*)pa, *(const v8bf*)(pa + 16));
        }
#pragma unroll
        for (int nt = 0; nt < 2; ++nt) {
            const unsigned short* pb = &sW[(wn + nt * 16 + lr) * LDS_STRIDE + half * 8];
            bf_[nt] = mkfrag(*(const v8bf*)pb, *(const v8bf*)(pb + 16));
        }
#pragma unroll
        for (int mt = 0; mt < 4; ++mt)
#pragma unroll
            for (int nt = 0; nt < 2; ++nt)
                acc[mt][nt] = wmma_bf16(af[mt], bf_[nt], acc[mt][nt]);
        __syncthreads();
    }

    // epilogue (C layout: row = v + 8*half, col = lane&15)
#pragma unroll
    for (int mt = 0; mt < 4; ++mt) {
#pragma unroll
        for (int nt = 0; nt < 2; ++nt) {
            int col = n0 + wn + nt * 16 + lr;
            float bv = bias ? bias[col] : 0.f;
#pragma unroll
            for (int v = 0; v < 8; ++v) {
                int row = m0 + wm + mt * 16 + (half ? 8 : 0) + v;
                size_t idx = (size_t)row * ldc + col;
                float x = acc[mt][nt][v] + bv;
                if (mode == 0) {
                    Cf[idx] = x;
                } else if (mode == 1) {
                    Cb[idx] = f2bfu(x);
                } else if (mode == 2) {
                    float g = 0.5f * x * (1.f + tanhf(0.7978845608f * (x + 0.044715f * x * x * x)));
                    Cb[idx] = f2bfu(g);
                } else { // mode 3: residual + gate*(x)
                    Cf[idx] = resid[idx] + gate[col] * x;
                }
            }
        }
    }
}

// ---------------------------------------------------------------------------
// 5) per-head RMSNorm + RoPE, f32 -> bf16
// ---------------------------------------------------------------------------
__global__ __launch_bounds__(128) void rope_rms_kernel(const float* __restrict__ X,
                                                       const float* __restrict__ rmsw,
                                                       const float* __restrict__ cosb,
                                                       const float* __restrict__ sinb,
                                                       unsigned short* __restrict__ Y) {
    int s = blockIdx.x / NHEAD, h = blockIdx.x % NHEAD;
    int d = threadIdx.x;
    const float* xr = X + (size_t)s * DIMD + h * HDIM;
    float x = xr[d];
    float ss = x * x;
#pragma unroll
    for (int m = 16; m > 0; m >>= 1) ss += __shfl_xor(ss, m, 32);
    __shared__ float red[4];
    if ((threadIdx.x & 31) == 0) red[threadIdx.x >> 5] = ss;
    __syncthreads();
    float tot = red[0] + red[1] + red[2] + red[3];
    float inv = rsqrtf(tot / (float)HDIM + 1e-6f);
    float xn = x * inv * rmsw[d];
    float xp = __shfl_xor(x, 1, 32) * inv * rmsw[d ^ 1];   // pair partner
    float c  = cosb[s * HDIM + d];
    float sn = sinb[s * HDIM + d];
    float rot = (d & 1) ? xp : -xp;     // [-x_odd, x_even] interleave
    Y[(size_t)s * DIMD + h * HDIM + d] = f2bfu(xn * c + rot * sn);
}

// ---------------------------------------------------------------------------
// 6) Flash attention via WMMA. block = 128 thr (4 waves), one (head, 64 qrows).
//    Per wave: 16 q rows, online softmax, K-chunks of 32 keys.
// ---------------------------------------------------------------------------
__global__ __launch_bounds__(128) void attn_kernel(const unsigned short* __restrict__ Q,
                                                   const unsigned short* __restrict__ Km,
                                                   const unsigned short* __restrict__ Vm,
                                                   float* __restrict__ O, int addFlag) {
    __shared__ __align__(16) unsigned short sK[32 * 136];   // [key][hd] (+pad)
    __shared__ __align__(16) unsigned short sVT[128 * 40];  // [hd][key] (+pad)
    __shared__ __align__(16) unsigned short sP[4][16 * 40]; // per-wave P tile

    const int h  = blockIdx.x % NHEAD;
    const int qb = blockIdx.x / NHEAD;
    const int tid = threadIdx.x, lane = tid & 31, wid = tid >> 5;
    const int half = lane >> 4, lr = lane & 15;
    const int qrow0 = qb * 64 + wid * 16;

    // Q fragments (A layout): 4 hd-chunks of 32
    v16bf qf[4];
#pragma unroll
    for (int kk = 0; kk < 4; ++kk) {
        const unsigned short* gq =
            Q + (size_t)(qrow0 + lr) * DIMD + h * HDIM + kk * 32 + half * 8;
        qf[kk] = mkfrag(*(const v8bf*)gq, *(const v8bf*)(gq + 16));
    }

    v8f o[8];
#pragma unroll
    for (int nt = 0; nt < 8; ++nt) o[nt] = (v8f)0.f;
    float rm[8], rl[8];
#pragma unroll
    for (int v = 0; v < 8; ++v) { rm[v] = -3.0e38f; rl[v] = 0.f; }

    const int krow = tid >> 2, seg = tid & 3;  // staging map: 32 rows x 4 segs
    const float scale = 0.08838834764831845f;  // 1/sqrt(128)

#pragma unroll 1
    for (int j0 = 0; j0 < S_LEN; j0 += 32) {
        // stage K rows [j0..j0+31] and V^T
#pragma unroll
        for (int i = 0; i < 4; ++i) {
            const unsigned short* gk =
                Km + (size_t)(j0 + krow) * DIMD + h * HDIM + seg * 32 + i * 8;
            *(v8bf*)&sK[krow * 136 + seg * 32 + i * 8] = *(const v8bf*)gk;
            const unsigned short* gv =
                Vm + (size_t)(j0 + krow) * DIMD + h * HDIM + seg * 32 + i * 8;
            v8bf vv = *(const v8bf*)gv;
#pragma unroll
            for (int e = 0; e < 8; ++e)
                sVT[(seg * 32 + i * 8 + e) * 40 + krow] =
                    __builtin_bit_cast(unsigned short, vv[e]);
        }
        __syncthreads();

        // scores: two 16x16 tiles (keys jt*16)
        v8f sc[2];
#pragma unroll
        for (int jt = 0; jt < 2; ++jt) {
            sc[jt] = (v8f)0.f;
#pragma unroll
            for (int kk = 0; kk < 4; ++kk) {
                const unsigned short* pb = &sK[(jt * 16 + lr) * 136 + kk * 32 + half * 8];
                v16bf bfr = mkfrag(*(const v8bf*)pb, *(const v8bf*)(pb + 16));
                sc[jt] = wmma_bf16(qf[kk], bfr, sc[jt]);
            }
        }
        // online softmax per row (row = v + 8*half; 16 lanes of a half share row)
#pragma unroll
        for (int v = 0; v < 8; ++v) {
            float s0 = sc[0][v] * scale, s1 = sc[1][v] * scale;
            float t = fmaxf(s0, s1);
#pragma unroll
            for (int m = 8; m > 0; m >>= 1) t = fmaxf(t, __shfl_xor(t, m, 16));
            float nm = fmaxf(rm[v], t);
            float alpha = __expf(rm[v] - nm);
            rm[v] = nm;
            float p0 = __expf(s0 - nm), p1 = __expf(s1 - nm);
            float rs = p0 + p1;
#pragma unroll
            for (int m = 8; m > 0; m >>= 1) rs += __shfl_xor(rs, m, 16);
            rl[v] = rl[v] * alpha + rs;
#pragma unroll
            for (int nt = 0; nt < 8; ++nt) o[nt][v] *= alpha;
            int prow = v + 8 * half;
            sP[wid][prow * 40 + lr]      = f2bfu(p0);
            sP[wid][prow * 40 + 16 + lr] = f2bfu(p1);
        }
        // P @ V  (A-frag from sP, B-frag from transposed V)
        const unsigned short* pp = &sP[wid][lr * 40 + half * 8];
        v16bf pf = mkfrag(*(const v8bf*)pp, *(const v8bf*)(pp + 16));
#pragma unroll
        for (int nt = 0; nt < 8; ++nt) {
            const unsigned short* pv = &sVT[(nt * 16 + lr) * 40 + half * 8];
            v16bf vf = mkfrag(*(const v8bf*)pv, *(const v8bf*)(pv + 16));
            o[nt] = wmma_bf16(pf, vf, o[nt]);
        }
        __syncthreads();
    }

    // write out: O[qrow][h*128 + col] = o / l
#pragma unroll
    for (int nt = 0; nt < 8; ++nt) {
#pragma unroll
        for (int v = 0; v < 8; ++v) {
            int row = qrow0 + v + 8 * half;
            int col = h * HDIM + nt * 16 + lr;
            size_t idx = (size_t)row * DIMD + col;
            float val = o[nt][v] / rl[v];
            if (addFlag) O[idx] += val; else O[idx] = val;
        }
    }
}

// ---------------------------------------------------------------------------
// 7) f32 attn buffer -> bf16 column block of concat buffer (ld = 15360)
// ---------------------------------------------------------------------------
__global__ __launch_bounds__(256) void cvt_attn(const float* __restrict__ A,
                                                unsigned short* __restrict__ C) {
    int i = blockIdx.x * 256 + threadIdx.x;
    int row = i / DIMD, col = i % DIMD;
    C[(size_t)row * CATD + col] = f2bfu(A[i]);
}

// ---------------------------------------------------------------------------
// host: orchestrate the whole block on `stream`
// ---------------------------------------------------------------------------
extern "C" void kernel_launch(void* const* d_in, const int* in_sizes, int n_in,
                              void* d_out, int out_size, void* d_ws, size_t ws_size,
                              hipStream_t stream) {
    const float* hs     = (const float*)d_in[0];
    const float* temb   = (const float*)d_in[1];
    const float* hsc    = (const float*)d_in[2];
    const float* tembc  = (const float*)d_in[3];
    const float* cosb   = (const float*)d_in[4];
    const float* sinb   = (const float*)d_in[5];
    const float* norm_w = (const float*)d_in[6];
    const float* norm_b = (const float*)d_in[7];
    const float* normc_w= (const float*)d_in[8];
    const float* normc_b= (const float*)d_in[9];
    const float* mlp_w  = (const float*)d_in[10];
    const float* mlp_b  = (const float*)d_in[11];
    const float* out_w  = (const float*)d_in[12];
    const float* out_b  = (const float*)d_in[13];
    const float* q_w    = (const float*)d_in[14];
    const float* q_b    = (const float*)d_in[15];
    const float* k_w    = (const float*)d_in[16];
    const float* k_b    = (const float*)d_in[17];
    const float* v_w    = (const float*)d_in[18];
    const float* v_b    = (const float*)d_in[19];
    const float* rms_q  = (const float*)d_in[20];
    const float* rms_k  = (const float*)d_in[21];
    const float* qc_w   = (const float*)d_in[22];
    const float* qc_b   = (const float*)d_in[23];
    const float* kc_w   = (const float*)d_in[24];
    const float* kc_b   = (const float*)d_in[25];
    const float* vc_w   = (const float*)d_in[26];
    const float* vc_b   = (const float*)d_in[27];
    const float* rms_qc = (const float*)d_in[28];
    const float* rms_kc = (const float*)d_in[29];

    const size_t SZ = (size_t)S_LEN * DIMD;

    // workspace carve-up
    float* silu_t = (float*)d_ws;                       // 2*3072
    float* e_buf  = silu_t + 2 * DIMD;                  // 2*9216
    unsigned short* nh   = (unsigned short*)(e_buf + 2 * 3 * DIMD);
    unsigned short* nh_c = nh + SZ;
    float* qf  = (float*)(nh_c + SZ);                   // 4 f32 tensors
    float* kf  = qf  + SZ;
    float* qcf = kf  + SZ;
    float* kcf = qcf + SZ;
    unsigned short* qb_  = (unsigned short*)(kcf + SZ); // 6 bf16 tensors
    unsigned short* kb_  = qb_  + SZ;
    unsigned short* qcb  = kb_  + SZ;
    unsigned short* kcb  = qcb  + SZ;
    unsigned short* vb_  = kcb  + SZ;
    unsigned short* vcb  = vb_  + SZ;
    float* attn_m = (float*)(vcb + SZ);
    float* attn_c = attn_m + SZ;
    unsigned short* cat_m = (unsigned short*)(attn_c + SZ); // [1024][15360]
    unsigned short* cat_c = cat_m + (size_t)S_LEN * CATD;

    float* out_h  = (float*)d_out;
    float* out_hc = out_h + SZ;

    // 1) silu(temb)
    silu_kernel<<<(2 * DIMD + 255) / 256, 256, 0, stream>>>(temb, tembc, silu_t);
    // 2) adaLN modulation vectors
    emb_gemv<<<(3 * DIMD) / 8, 256, 0, stream>>>(silu_t, norm_w, norm_b, e_buf);
    emb_gemv<<<(3 * DIMD) / 8, 256, 0, stream>>>(silu_t + DIMD, normc_w, normc_b, e_buf + 3 * DIMD);
    // 3) adaLN-zero norm -> bf16
    adanorm_kernel<<<S_LEN, 256, 0, stream>>>(hs,  e_buf,            nh);
    adanorm_kernel<<<S_LEN, 256, 0, stream>>>(hsc, e_buf + 3 * DIMD, nh_c);

    // 4) MLP (gelu fused) -> concat[:,3072:]
    dim3 gm(MLPD / 128, S_LEN / 128);
    gemm_bf16<<<gm, 256, 0, stream>>>(nh,   DIMD, mlp_w, DIMD, mlp_b, nullptr, cat_m + DIMD, CATD, DIMD, 2, nullptr, nullptr);
    gemm_bf16<<<gm, 256, 0, stream>>>(nh_c, DIMD, mlp_w, DIMD, mlp_b, nullptr, cat_c + DIMD, CATD, DIMD, 2, nullptr, nullptr);

    // 5) QKV projections
    dim3 gq(DIMD / 128, S_LEN / 128);
    gemm_bf16<<<gq, 256, 0, stream>>>(nh,   DIMD, q_w,  DIMD, q_b,  qf,  nullptr, DIMD, DIMD, 0, nullptr, nullptr);
    gemm_bf16<<<gq, 256, 0, stream>>>(nh,   DIMD, k_w,  DIMD, k_b,  kf,  nullptr, DIMD, DIMD, 0, nullptr, nullptr);
    gemm_bf16<<<gq, 256, 0, stream>>>(nh,   DIMD, v_w,  DIMD, v_b,  nullptr, vb_, DIMD, DIMD, 1, nullptr, nullptr);
    gemm_bf16<<<gq, 256, 0, stream>>>(nh_c, DIMD, qc_w, DIMD, qc_b, qcf, nullptr, DIMD, DIMD, 0, nullptr, nullptr);
    gemm_bf16<<<gq, 256, 0, stream>>>(nh_c, DIMD, kc_w, DIMD, kc_b, kcf, nullptr, DIMD, DIMD, 0, nullptr, nullptr);
    gemm_bf16<<<gq, 256, 0, stream>>>(nh_c, DIMD, vc_w, DIMD, vc_b, nullptr, vcb, DIMD, DIMD, 1, nullptr, nullptr);

    // 6) RMS + RoPE -> bf16
    int rblk = S_LEN * NHEAD;
    rope_rms_kernel<<<rblk, 128, 0, stream>>>(qf,  rms_q,  cosb, sinb, qb_);
    rope_rms_kernel<<<rblk, 128, 0, stream>>>(kf,  rms_k,  cosb, sinb, kb_);
    rope_rms_kernel<<<rblk, 128, 0, stream>>>(qcf, rms_qc, cosb, sinb, qcb);
    rope_rms_kernel<<<rblk, 128, 0, stream>>>(kcf, rms_kc, cosb, sinb, kcb);

    // 7) attention: main, cross (accumulated), control
    int ablk = NHEAD * (S_LEN / 64);
    attn_kernel<<<ablk, 128, 0, stream>>>(qb_, kb_,  vb_, attn_m, 0);
    attn_kernel<<<ablk, 128, 0, stream>>>(qb_, kcb,  vcb, attn_m, 1);
    attn_kernel<<<ablk, 128, 0, stream>>>(qcb, kcb,  vcb, attn_c, 0);

    // 8) attn f32 -> bf16 into concat[:, :3072]
    cvt_attn<<<(int)(SZ / 256), 256, 0, stream>>>(attn_m, cat_m);
    cvt_attn<<<(int)(SZ / 256), 256, 0, stream>>>(attn_c, cat_c);

    // 9) out projection with fused gate*(x)+residual
    dim3 go(DIMD / 128, S_LEN / 128);
    gemm_bf16<<<go, 256, 0, stream>>>(cat_m, CATD, out_w, CATD, out_b, out_h,  nullptr, DIMD, CATD, 3,
                                      e_buf + 2 * DIMD, hs);
    gemm_bf16<<<go, 256, 0, stream>>>(cat_c, CATD, out_w, CATD, out_b, out_hc, nullptr, DIMD, CATD, 3,
                                      e_buf + 3 * DIMD + 2 * DIMD, hsc);
}
